// TransformerDecoderBlock_56564719289048
// MI455X (gfx1250) — matrix operand order repaired and checked by reading, not compiled, not run on hardware
//
#include <hip/hip_runtime.h>
#include <hip/hip_bf16.h>
#include <math.h>

typedef __attribute__((ext_vector_type(16))) _Float16 v16h;
typedef __attribute__((ext_vector_type(8)))  _Float16 v8h;
typedef __attribute__((ext_vector_type(8)))  float    v8f;
typedef __attribute__((ext_vector_type(4)))  int      v4i;

#define B_   2
#define S_   2048
#define D_   1024
#define HQ_  16
#define HKV_ 4
#define HD_  64
#define P_   4096
#define NEGBIG (-1.0e9f)

#define LDS_AS __attribute__((address_space(3)))
#define GLB_AS __attribute__((address_space(1)))

// Build an A-fragment (16x32 f16, wave32 layout) from two contiguous 8-half chunks.
// Lane l (m = l%16, g = l/16): element e -> K = e + 16*g (callers pass hi8 = lo8 + 8).
// Also used for B-fragments (element e -> K = e + 16*g) with hi8 = lo8 + 8.
__device__ inline v16h make_frag(const _Float16* lo8, const _Float16* hi8) {
  v8h lo = *(const v8h*)lo8;
  v8h hi = *(const v8h*)hi8;
  v16h a;
#pragma unroll
  for (int i = 0; i < 8; ++i) { a[i] = lo[i]; a[i + 8] = hi[i]; }
  return a;
}

__device__ inline v8f wmma_f16(v16h a, v16h b, v8f c) {
  return __builtin_amdgcn_wmma_f32_16x16x32_f16(false, a, false, b, (short)0, c,
                                                false, false);
}

// ---- CDNA5 async global->LDS copy (ASYNCcnt-tracked), with asm fallback ----
__device__ inline void async_copy16(const _Float16* g, _Float16* l) {
#if __has_builtin(__builtin_amdgcn_global_load_async_to_lds_b128)
  __builtin_amdgcn_global_load_async_to_lds_b128(
      (GLB_AS v4i*)(const void*)g, (LDS_AS v4i*)(void*)l, 0, 0);
#else
  unsigned lofs = (unsigned)(unsigned long long)(LDS_AS char*)(void*)l;
  asm volatile("global_load_async_to_lds_b128 %0, %1, off"
               :: "v"(lofs), "v"((unsigned long long)g)
               : "memory");
#endif
}

__device__ inline void async_wait0() {
#if __has_builtin(__builtin_amdgcn_s_wait_asynccnt)
  __builtin_amdgcn_s_wait_asynccnt(0);
#else
  asm volatile("s_wait_asynccnt 0" ::: "memory");
#endif
}

// ---------------------------------------------------------------------------
// Weight convert + transpose: w[K][N] f32 -> wt[N][K] f16
// ---------------------------------------------------------------------------
__global__ __launch_bounds__(256) void cvt_transpose(const float* __restrict__ w,
                                                     _Float16* __restrict__ wt,
                                                     int K, int N) {
  int idx = blockIdx.x * 256 + threadIdx.x;
  if (idx >= K * N) return;
  int kk = idx / N;
  int n  = idx - kk * N;
  wt[(size_t)n * K + kk] = (_Float16)w[(size_t)kk * N + n];
}

// ---------------------------------------------------------------------------
// LayerNorm (f32 in, f16 out). One block per row of D_=1024, 256 threads.
// ---------------------------------------------------------------------------
__global__ __launch_bounds__(256) void layernorm_f16(const float* __restrict__ x,
                                                     const float* __restrict__ gam,
                                                     const float* __restrict__ bet,
                                                     _Float16* __restrict__ out) {
  __shared__ float s_sum[256];
  __shared__ float s_sq[256];
  const int row = blockIdx.x;
  const int t = threadIdx.x;
  const float* xr = x + (size_t)row * D_;
  float v0 = xr[t], v1 = xr[t + 256], v2 = xr[t + 512], v3 = xr[t + 768];
  s_sum[t] = v0 + v1 + v2 + v3;
  s_sq[t]  = v0 * v0 + v1 * v1 + v2 * v2 + v3 * v3;
  __syncthreads();
  for (int off = 128; off > 0; off >>= 1) {
    if (t < off) { s_sum[t] += s_sum[t + off]; s_sq[t] += s_sq[t + off]; }
    __syncthreads();
  }
  const float mean = s_sum[0] * (1.0f / D_);
  const float var  = s_sq[0] * (1.0f / D_) - mean * mean;
  const float rstd = rsqrtf(var + 1e-5f);
  _Float16* orow = out + (size_t)row * D_;
  orow[t]       = (_Float16)((v0 - mean) * rstd * gam[t]       + bet[t]);
  orow[t + 256] = (_Float16)((v1 - mean) * rstd * gam[t + 256] + bet[t + 256]);
  orow[t + 512] = (_Float16)((v2 - mean) * rstd * gam[t + 512] + bet[t + 512]);
  orow[t + 768] = (_Float16)((v3 - mean) * rstd * gam[t + 768] + bet[t + 768]);
}

// ---------------------------------------------------------------------------
// WMMA GEMM: C[M,N] = A[M,K](f16,row) @ Bt[N,K](f16, B transposed) + epilogue.
// 256 threads = 8 waves; wave tile 32x64; block tile 128x128.
// B tile (128 x 64 halves) staged in LDS via async copies, double buffered,
// LDS row stride padded to 72 halves for conflict-free ds_load_b128.
// ---------------------------------------------------------------------------
enum { EPI_PLAIN = 0, EPI_QHEAD = 1, EPI_KHEAD = 2, EPI_VT = 3,
       EPI_RESID = 4, EPI_BIAS_GELU = 5, EPI_BIAS_RESID = 6 };

#define BSTRIDE 72  // padded LDS row stride (halves)

template <int EPI>
__global__ __launch_bounds__(256) void gemm_wmma(
    const _Float16* __restrict__ A, const _Float16* __restrict__ Bt,
    void* __restrict__ Out, const float* __restrict__ bias,
    const float* __restrict__ resid, int M, int N, int K) {
  __shared__ __align__(16) _Float16 bsh[2][128 * BSTRIDE];
  const int lane = threadIdx.x & 31;
  const int wave = threadIdx.x >> 5;
  const int lh = lane & 15;
  const int g  = lane >> 4;
  const int m0 = blockIdx.y * 128 + (wave >> 1) * 32;
  const int nw = (wave & 1) * 64;           // wave's N offset inside block tile
  const int n0blk = blockIdx.x * 128;

  // async-stage one 128x64 B tile (each thread copies 4 x 16B, coalesced)
  auto issue_stage = [&](int kb, int buf) {
#pragma unroll
    for (int j = 0; j < 4; ++j) {
      const int c   = threadIdx.x + 256 * j;   // 0..1023
      const int row = c >> 3;
      const int col = (c & 7) * 8;
      async_copy16(Bt + (size_t)(n0blk + row) * K + kb + col,
                   &bsh[buf][row * BSTRIDE + col]);
    }
  };

  v8f acc[2][4];
#pragma unroll
  for (int mi = 0; mi < 2; ++mi)
#pragma unroll
    for (int ni = 0; ni < 4; ++ni) acc[mi][ni] = {};

  issue_stage(0, 0);
  int buf = 0;
  for (int kb = 0; kb < K; kb += 64) {
    async_wait0();
    __syncthreads();
    if (kb + 64 < K) issue_stage(kb + 64, buf ^ 1);
#pragma unroll
    for (int kk = 0; kk < 64; kk += 32) {
      v16h af[2];
#pragma unroll
      for (int mi = 0; mi < 2; ++mi) {
        const _Float16* ap = A + (size_t)(m0 + mi * 16 + lh) * K + kb + kk;
        af[mi] = make_frag(ap + 16 * g, ap + 16 * g + 8);
      }
      v16h bf[4];
#pragma unroll
      for (int ni = 0; ni < 4; ++ni) {
        const _Float16* bp = &bsh[buf][(nw + ni * 16 + lh) * BSTRIDE + kk + 16 * g];
        bf[ni] = make_frag(bp, bp + 8);
      }
#pragma unroll
      for (int mi = 0; mi < 2; ++mi)
#pragma unroll
        for (int ni = 0; ni < 4; ++ni)
          acc[mi][ni] = wmma_f16(af[mi], bf[ni], acc[mi][ni]);
    }
    __syncthreads();
    buf ^= 1;
  }

#pragma unroll
  for (int mi = 0; mi < 2; ++mi) {
#pragma unroll
    for (int ni = 0; ni < 4; ++ni) {
#pragma unroll
      for (int r = 0; r < 8; ++r) {
        const int row = m0 + mi * 16 + 8 * g + r;
        const int col = n0blk + nw + ni * 16 + lh;
        float v = acc[mi][ni][r];
        if (EPI == EPI_PLAIN) {
          ((float*)Out)[(size_t)row * N + col] = v;
        } else if (EPI == EPI_QHEAD) {
          const int b = row / S_, t = row - b * S_;
          const int h = col / HD_, hd = col - h * HD_;
          ((float*)Out)[(((size_t)b * HQ_ + h) * S_ + t) * HD_ + hd] = v;
        } else if (EPI == EPI_KHEAD) {
          const int b = row / S_, t = row - b * S_;
          const int h = col / HD_, hd = col - h * HD_;
          ((float*)Out)[(((size_t)b * HKV_ + h) * S_ + t) * HD_ + hd] = v;
        } else if (EPI == EPI_VT) {
          const int b = row / S_, t = row - b * S_;
          const int h = col / HD_, hd = col - h * HD_;
          ((_Float16*)Out)[(((size_t)b * HKV_ + h) * HD_ + hd) * S_ + t] =
              (_Float16)v;
        } else if (EPI == EPI_RESID) {
          ((float*)Out)[(size_t)row * N + col] = v + resid[(size_t)row * N + col];
        } else if (EPI == EPI_BIAS_GELU) {
          float u = v + bias[col];
          float gl = 0.5f * u *
                     (1.0f + tanhf(0.7978845608028654f * (u + 0.044715f * u * u * u)));
          ((_Float16*)Out)[(size_t)row * N + col] = (_Float16)gl;
        } else if (EPI == EPI_BIAS_RESID) {
          ((float*)Out)[(size_t)row * N + col] =
              v + bias[col] + resid[(size_t)row * N + col];
        }
      }
    }
  }
}

// ---------------------------------------------------------------------------
// RoPE: f32 head-major [B,H,S,64] -> f16 same layout.
// ---------------------------------------------------------------------------
__global__ __launch_bounds__(256) void rope_kernel(const float* __restrict__ in,
                                                   _Float16* __restrict__ out,
                                                   int H) {
  int idx = blockIdx.x * 256 + threadIdx.x;
  const int total = B_ * H * S_ * 32;
  if (idx >= total) return;
  const int i  = idx & 31;
  const int t  = (idx >> 5) % S_;
  const int bh = idx / (32 * S_);
  const float inv = powf(10000.0f, -(float)i / 32.0f);
  const float ang = (float)t * inv;
  const float cs = cosf(ang), sn = sinf(ang);
  const size_t base = ((size_t)bh * S_ + t) * HD_;
  const float x1 = in[base + i], x2 = in[base + 32 + i];
  out[base + i]      = (_Float16)(x1 * cs - x2 * sn);
  out[base + 32 + i] = (_Float16)(x1 * sn + x2 * cs);
}

// ---------------------------------------------------------------------------
// Flash attention. 128 threads = 4 waves; one wave per 16-query-row block.
// q: [B,HQ,S,64] f16 (roped), k: [B,HKV,S,64] f16 (roped), vt: [B,HKV,64,S] f16
// o: [B,S,D] f16.  Causal mask applied as +(-1e9) like the reference.
// ---------------------------------------------------------------------------
__global__ __launch_bounds__(128) void attn_kernel(const _Float16* __restrict__ q,
                                                   const _Float16* __restrict__ k,
                                                   const _Float16* __restrict__ vt,
                                                   _Float16* __restrict__ o) {
  __shared__ __align__(32) _Float16 plds[4][16 * 32];
  const int lane = threadIdx.x & 31;
  const int wave = threadIdx.x >> 5;
  const int lh = lane & 15;
  const int g  = lane >> 4;
  const int gw = blockIdx.x * 4 + wave;
  const int nmb  = S_ / 16;  // 128
  const int mblk = gw % nmb;
  const int h    = (gw / nmb) % HQ_;
  const int b    = gw / (nmb * HQ_);
  const int kvh  = h / (HQ_ / HKV_);
  const _Float16* qp = q  + ((size_t)b * HQ_  + h)   * S_  * HD_;
  const _Float16* kp = k  + ((size_t)b * HKV_ + kvh) * S_  * HD_;
  const _Float16* vp = vt + ((size_t)b * HKV_ + kvh) * HD_ * S_;
  const int m0 = mblk * 16;

  // Q fragments for K-dim 0..31 and 32..63.
  v16h qf[2];
  {
    const _Float16* ap = qp + (size_t)(m0 + lh) * HD_;
    qf[0] = make_frag(ap + 16 * g,      ap + 8 + 16 * g);
    qf[1] = make_frag(ap + 32 + 16 * g, ap + 40 + 16 * g);
  }

  float m_i[8], l_i[8];
  v8f oacc[4];
#pragma unroll
  for (int r = 0; r < 8; ++r) { m_i[r] = -1e30f; l_i[r] = 0.0f; }
#pragma unroll
  for (int ni = 0; ni < 4; ++ni) oacc[ni] = {};

  _Float16* pl = plds[wave];

  for (int cb = 0; cb <= m0 + 15; cb += 32) {
    // --- scores: 16 rows x 32 keys, K-reduce over 64 head dims ---
    v8f sc0 = {}, sc1 = {};
#pragma unroll
    for (int kd = 0; kd < 2; ++kd) {
      v16h kf0 = *(const v16h*)(kp + (size_t)(cb + lh) * HD_      + kd * 32 + 16 * g);
      v16h kf1 = *(const v16h*)(kp + (size_t)(cb + 16 + lh) * HD_ + kd * 32 + 16 * g);
      sc0 = wmma_f16(qf[kd], kf0, sc0);
      sc1 = wmma_f16(qf[kd], kf1, sc1);
    }
    float s0[8], s1[8], nm[8];
#pragma unroll
    for (int r = 0; r < 8; ++r) {
      const int row = m0 + 8 * g + r;
      const int c0 = cb + lh, c1 = cb + 16 + lh;
      s0[r] = sc0[r] * 0.125f + ((c0 <= row) ? 0.0f : NEGBIG);
      s1[r] = sc1[r] * 0.125f + ((c1 <= row) ? 0.0f : NEGBIG);
      nm[r] = fmaxf(s0[r], s1[r]);
    }
#pragma unroll
    for (int r = 0; r < 8; ++r)
#pragma unroll
      for (int mk = 1; mk < 16; mk <<= 1)
        nm[r] = fmaxf(nm[r], __shfl_xor(nm[r], mk, 32));
    float alpha[8], rs[8];
#pragma unroll
    for (int r = 0; r < 8; ++r) {
      const float mnew = fmaxf(m_i[r], nm[r]);
      alpha[r] = __expf(m_i[r] - mnew);
      m_i[r] = mnew;
      s0[r] = __expf(s0[r] - mnew);
      s1[r] = __expf(s1[r] - mnew);
      rs[r] = s0[r] + s1[r];
    }
#pragma unroll
    for (int r = 0; r < 8; ++r)
#pragma unroll
      for (int mk = 1; mk < 16; mk <<= 1)
        rs[r] += __shfl_xor(rs[r], mk, 32);
#pragma unroll
    for (int r = 0; r < 8; ++r) l_i[r] = l_i[r] * alpha[r] + rs[r];
#pragma unroll
    for (int ni = 0; ni < 4; ++ni)
#pragma unroll
      for (int r = 0; r < 8; ++r) oacc[ni][r] *= alpha[r];

    // --- P (C-frag layout) -> LDS -> A-frag layout ---
#pragma unroll
    for (int r = 0; r < 8; ++r) {
      const int prow = 8 * g + r;
      pl[prow * 32 + lh]      = (_Float16)s0[r];
      pl[prow * 32 + 16 + lh] = (_Float16)s1[r];
    }
    v16h pa = make_frag(pl + lh * 32 + 16 * g, pl + lh * 32 + 8 + 16 * g);

    // --- O += P @ V (V transposed: rows = head dims, contiguous in keys) ---
#pragma unroll
    for (int ni = 0; ni < 4; ++ni) {
      v16h vf = *(const v16h*)(vp + (size_t)(ni * 16 + lh) * S_ + cb + 16 * g);
      oacc[ni] = wmma_f16(pa, vf, oacc[ni]);
    }
  }

  _Float16* op = o + (size_t)b * S_ * D_ + h * HD_;
#pragma unroll
  for (int ni = 0; ni < 4; ++ni)
#pragma unroll
    for (int r = 0; r < 8; ++r) {
      const int t  = m0 + 8 * g + r;
      const int hd = ni * 16 + lh;
      op[(size_t)t * D_ + hd] = (_Float16)(oacc[ni][r] / l_i[r]);
    }
}

// ---------------------------------------------------------------------------
// Host-side launch
// ---------------------------------------------------------------------------
extern "C" void kernel_launch(void* const* d_in, const int* in_sizes, int n_in,
                              void* d_out, int out_size, void* d_ws,
                              size_t ws_size, hipStream_t stream) {
  const float* x    = (const float*)d_in[0];
  const float* wq   = (const float*)d_in[2];
  const float* wk   = (const float*)d_in[3];
  const float* wv   = (const float*)d_in[4];
  const float* wo   = (const float*)d_in[5];
  const float* ln1g = (const float*)d_in[6];
  const float* ln1b = (const float*)d_in[7];
  const float* ln2g = (const float*)d_in[8];
  const float* ln2b = (const float*)d_in[9];
  const float* w1   = (const float*)d_in[10];
  const float* b1   = (const float*)d_in[11];
  const float* w2   = (const float*)d_in[12];
  const float* b2   = (const float*)d_in[13];
  float* out = (float*)d_out;

  char* wsp = (char*)d_ws;
  auto alloc = [&](size_t bytes) -> char* {
    char* p = wsp;
    wsp += (bytes + 255) & ~(size_t)255;
    return p;
  };
  const size_t BSD = (size_t)B_ * S_ * D_;
  _Float16* h16   = (_Float16*)alloc(BSD * 2);
  _Float16* wq_t  = (_Float16*)alloc((size_t)D_ * D_ * 2);
  _Float16* wk_t  = (_Float16*)alloc((size_t)D_ * HKV_ * HD_ * 2);
  _Float16* wv_t  = (_Float16*)alloc((size_t)D_ * HKV_ * HD_ * 2);
  _Float16* wo_t  = (_Float16*)alloc((size_t)D_ * D_ * 2);
  _Float16* w1_t  = (_Float16*)alloc((size_t)D_ * P_ * 2);
  _Float16* w2_t  = (_Float16*)alloc((size_t)P_ * D_ * 2);
  float*    qf32  = (float*)alloc(BSD * 4);
  float*    kf32  = (float*)alloc((size_t)B_ * HKV_ * S_ * HD_ * 4);
  _Float16* q16   = (_Float16*)alloc(BSD * 2);
  _Float16* k16   = (_Float16*)alloc((size_t)B_ * HKV_ * S_ * HD_ * 2);
  _Float16* vt16  = (_Float16*)alloc((size_t)B_ * HKV_ * S_ * HD_ * 2);
  _Float16* o16   = (_Float16*)alloc(BSD * 2);
  float*    x2    = (float*)alloc(BSD * 4);
  _Float16* h2_16 = (_Float16*)alloc(BSD * 2);
  _Float16* act16 = (_Float16*)alloc((size_t)B_ * S_ * P_ * 2);

  const int M = B_ * S_;  // 4096

  // 1. weight conversion (fp32 -> f16, transposed)
  cvt_transpose<<<(D_ * D_) / 256, 256, 0, stream>>>(wq, wq_t, D_, D_);
  cvt_transpose<<<(D_ * HKV_ * HD_) / 256, 256, 0, stream>>>(wk, wk_t, D_, HKV_ * HD_);
  cvt_transpose<<<(D_ * HKV_ * HD_) / 256, 256, 0, stream>>>(wv, wv_t, D_, HKV_ * HD_);
  cvt_transpose<<<(D_ * D_) / 256, 256, 0, stream>>>(wo, wo_t, D_, D_);
  cvt_transpose<<<(D_ * P_) / 256, 256, 0, stream>>>(w1, w1_t, D_, P_);
  cvt_transpose<<<(P_ * D_) / 256, 256, 0, stream>>>(w2, w2_t, P_, D_);

  // 2. LN1
  layernorm_f16<<<M, 256, 0, stream>>>(x, ln1g, ln1b, h16);

  // 3. QKV projections (WMMA + async LDS staging)
  gemm_wmma<EPI_QHEAD><<<dim3(D_ / 128, M / 128), 256, 0, stream>>>(
      h16, wq_t, qf32, nullptr, nullptr, M, D_, D_);
  gemm_wmma<EPI_KHEAD><<<dim3((HKV_ * HD_) / 128, M / 128), 256, 0, stream>>>(
      h16, wk_t, kf32, nullptr, nullptr, M, HKV_ * HD_, D_);
  gemm_wmma<EPI_VT><<<dim3((HKV_ * HD_) / 128, M / 128), 256, 0, stream>>>(
      h16, wv_t, vt16, nullptr, nullptr, M, HKV_ * HD_, D_);

  // 4. RoPE
  rope_kernel<<<(B_ * HQ_ * S_ * 32) / 256, 256, 0, stream>>>(qf32, q16, HQ_);
  rope_kernel<<<(B_ * HKV_ * S_ * 32) / 256, 256, 0, stream>>>(kf32, k16, HKV_);

  // 5. Attention (flash, WMMA)
  attn_kernel<<<(B_ * HQ_ * (S_ / 16)) / 4, 128, 0, stream>>>(q16, k16, vt16, o16);

  // 6. Output projection + residual
  gemm_wmma<EPI_RESID><<<dim3(D_ / 128, M / 128), 256, 0, stream>>>(
      o16, wo_t, x2, nullptr, x, M, D_, D_);

  // 7. LN2
  layernorm_f16<<<M, 256, 0, stream>>>(x2, ln2g, ln2b, h2_16);

  // 8. MLP
  gemm_wmma<EPI_BIAS_GELU><<<dim3(P_ / 128, M / 128), 256, 0, stream>>>(
      h2_16, w1_t, act16, b1, nullptr, M, P_, D_);
  gemm_wmma<EPI_BIAS_RESID><<<dim3(D_ / 128, M / 128), 256, 0, stream>>>(
      act16, w2_t, out, b2, x2, M, D_, P_);
}
